// GatedDeltaNet_55173149885063
// MI455X (gfx1250) — matrix operand order repaired and checked by reading, not compiled, not run on hardware
//
#include <hip/hip_runtime.h>

// ---- problem constants (match reference) ----
#define BB   2
#define SS   2048
#define DD   1024
#define HVn  16
#define HKn  8
#define DKc  64
#define DVc  64
#define KSc  4
#define KDIMc 512
#define VDIMc 1024
#define CDIMc 2048
#define MMr  (BB*SS)      // 4096 tokens
#define EPSc 1e-6f

typedef __attribute__((ext_vector_type(16))) __bf16 v16bf;
typedef __attribute__((ext_vector_type(8)))  float  v8f;

__device__ __forceinline__ __bf16 f2bf(float f) {
  unsigned u = __float_as_uint(f);
  u += 0x7FFFu + ((u >> 16) & 1u);          // round-to-nearest-even
  unsigned short h = (unsigned short)(u >> 16);
  return __builtin_bit_cast(__bf16, h);
}

// ---------------- cast f32 -> bf16 (4 elems / thread) ----------------
__global__ __launch_bounds__(256) void cast_f32_bf16(const float* __restrict__ src,
                                                     __bf16* __restrict__ dst, int n4) {
  int i = blockIdx.x * 256 + threadIdx.x;
  if (i >= n4) return;
  float4 f = ((const float4*)src)[i];
  dst[i*4+0] = f2bf(f.x); dst[i*4+1] = f2bf(f.y);
  dst[i*4+2] = f2bf(f.z); dst[i*4+3] = f2bf(f.w);
}

// ---------------- beta / g projections (tiny dots, HV=16) ----------------
__global__ __launch_bounds__(32) void proj_bg_kernel(const float* __restrict__ x,
    const float* __restrict__ Wb, const float* __restrict__ Wa,
    const float* __restrict__ dtb, const float* __restrict__ Alog,
    float* __restrict__ beta, float* __restrict__ gbuf) {
  int row = blockIdx.x;              // token
  int t = threadIdx.x;               // 0..31 : 16 beta dots + 16 g dots
  int h = t & 15;
  const float4* xr = (const float4*)(x + (size_t)row * DD);
  const float4* w  = (const float4*)(((t < 16) ? Wb : Wa) + (size_t)h * DD);
  float acc = 0.f;
  for (int i = 0; i < DD/4; ++i) {
    float4 a = xr[i], c = w[i];
    acc += a.x*c.x + a.y*c.y + a.z*c.z + a.w*c.w;
  }
  if (t < 16) {
    beta[(size_t)row * HVn + h] = 1.f / (1.f + __expf(-acc));
  } else {
    float u  = acc + dtb[h];
    float sp = (u > 20.f) ? u : log1pf(__expf(u));   // softplus
    gbuf[(size_t)row * HVn + h] = -__expf(Alog[h]) * sp;
  }
}

// ---------------- WMMA bf16 GEMM:  C[M,N] = A[M,K] * W[N,K]^T ----------------
__device__ __forceinline__ v16bf load_a_frag(const __bf16* __restrict__ A, int K,
                                             int row0, int k0, int lane) {
  int half = lane >> 4, m = lane & 15;
  const __bf16* p = A + (size_t)(row0 + m) * K + k0;
  v16bf a;
  ((float4*)&a)[0] = *(const float4*)(p + half * 8);        // K = half*8 .. +7
  ((float4*)&a)[1] = *(const float4*)(p + 16 + half * 8);   // K = 16+half*8 .. +7
  return a;
}
__device__ __forceinline__ v16bf load_b_frag(const __bf16* __restrict__ W, int K,
                                             int col0, int k0, int lane) {
  int half = lane >> 4, n = lane & 15;
  const __bf16* p = W + (size_t)(col0 + n) * K + k0;
  v16bf b;
  ((float4*)&b)[0] = *(const float4*)(p + half * 8);
  ((float4*)&b)[1] = *(const float4*)(p + 16 + half * 8);
  return b;
}
__device__ __forceinline__ void store_c(float* __restrict__ C, int N,
                                        int row0, int col0, int lane, v8f acc) {
  int half = lane >> 4, n = lane & 15;
  float* cp = C + (size_t)(row0 + half * 8) * N + col0 + n;
  #pragma unroll
  for (int r = 0; r < 8; ++r) cp[(size_t)r * N] = acc[r];
}

__global__ __launch_bounds__(256) void wmma_gemm_bf16(const __bf16* __restrict__ A,
    const __bf16* __restrict__ W, float* __restrict__ C, int M, int N, int K) {
  int lane = threadIdx.x & 31;
  int wid  = threadIdx.x >> 5;                 // 8 waves: 2 (m) x 4 (n)
  int row0 = blockIdx.x * 64  + (wid & 1) * 32;
  int col0 = blockIdx.y * 128 + (wid >> 1) * 32;
  v8f a00 = {}, a01 = {}, a10 = {}, a11 = {};
  for (int k0 = 0; k0 < K; k0 += 32) {
    if (k0 + 64 <= K) {   // prefetch next K-block (global_prefetch_b8)
      __builtin_prefetch((const void*)(A + (size_t)(row0 + (lane & 15)) * K + k0 + 32), 0, 0);
      __builtin_prefetch((const void*)(W + (size_t)(col0 + (lane & 15)) * K + k0 + 32), 0, 0);
    }
    v16bf fa0 = load_a_frag(A, K, row0,      k0, lane);
    v16bf fa1 = load_a_frag(A, K, row0 + 16, k0, lane);
    v16bf fb0 = load_b_frag(W, K, col0,      k0, lane);
    v16bf fb1 = load_b_frag(W, K, col0 + 16, k0, lane);
    a00 = __builtin_amdgcn_wmma_f32_16x16x32_bf16(false, fa0, false, fb0, (short)0, a00, false, false);
    a01 = __builtin_amdgcn_wmma_f32_16x16x32_bf16(false, fa0, false, fb1, (short)0, a01, false, false);
    a10 = __builtin_amdgcn_wmma_f32_16x16x32_bf16(false, fa1, false, fb0, (short)0, a10, false, false);
    a11 = __builtin_amdgcn_wmma_f32_16x16x32_bf16(false, fa1, false, fb1, (short)0, a11, false, false);
  }
  store_c(C, N, row0,      col0,      lane, a00);
  store_c(C, N, row0,      col0 + 16, lane, a01);
  store_c(C, N, row0 + 16, col0,      lane, a10);
  store_c(C, N, row0 + 16, col0 + 16, lane, a11);
}

// ---------------- depthwise causal conv (KS=4) + SiLU ----------------
__global__ __launch_bounds__(256) void conv_silu_kernel(const float* __restrict__ qkv,
    const float* __restrict__ cw, float* __restrict__ mixed) {
  size_t idx = (size_t)blockIdx.x * 256 + threadIdx.x;   // over MM*CDIM
  int c = (int)(idx % CDIMc);
  int row = (int)(idx / CDIMc);
  int s = row % SS;
  float acc = 0.f;
  #pragma unroll
  for (int j = 0; j < KSc; ++j) {
    int sp = s - (KSc - 1) + j;
    if (sp >= 0) acc += qkv[idx + (size_t)(j - (KSc - 1)) * CDIMc] * cw[c * KSc + j];
  }
  mixed[idx] = acc / (1.f + __expf(-acc));   // silu
}

// ---------------- l2norm q/k + GQA-ready layouts + v transpose ----------------
__global__ __launch_bounds__(256) void scanprep_kernel(const float* __restrict__ mixed,
    float* __restrict__ qn, float* __restrict__ kn, float* __restrict__ vt) {
  int row = blockIdx.x;                 // b*S + s
  int b = row / SS, s = row % SS;
  int t = threadIdx.x;
  int d = t & 63;
  __shared__ float ss[16];
  if (t < 16) ss[t] = 0.f;
  __syncthreads();
  const float* mrow = mixed + (size_t)row * CDIMc;
  float vals[4];
  #pragma unroll
  for (int p = 0; p < 4; ++p) {
    int gh = p * 4 + (t >> 6);          // 0..7 = q heads, 8..15 = k heads
    float v = (gh < 8) ? mrow[gh * DKc + d] : mrow[KDIMc + (gh - 8) * DKc + d];
    vals[p] = v;
    atomicAdd(&ss[gh], v * v);
  }
  __syncthreads();
  #pragma unroll
  for (int p = 0; p < 4; ++p) {
    int gh = p * 4 + (t >> 6);
    float inv = rsqrtf(ss[gh] + 1e-6f);
    if (gh < 8)
      qn[(((size_t)(b * HKn + gh)) * SS + s) * DKc + d] = vals[p] * inv * 0.125f; // * DK^-0.5
    else
      kn[(((size_t)(b * HKn + (gh - 8))) * SS + s) * DKc + d] = vals[p] * inv;
  }
  #pragma unroll
  for (int p = 0; p < 4; ++p) {         // v: [B,S,HV,DV] -> [B,HV,S,DV]
    int idx = p * 256 + t;
    int hv = idx >> 6, dd = idx & 63;
    vt[(((size_t)(b * HVn + hv)) * SS + s) * DVc + dd] = mrow[2 * KDIMc + idx];
  }
}

// ---------------- sequential gated delta-rule scan ----------------
// One block per (b, hv); 64 threads; thread t owns state column v = t (64 f32 in VGPRs).
__global__ __launch_bounds__(64) void scan_kernel(const float* __restrict__ qn,
    const float* __restrict__ kn, const float* __restrict__ vt,
    const float* __restrict__ gbuf, const float* __restrict__ bbuf,
    float* __restrict__ obuf) {
  int bh = blockIdx.x;
  int b = bh / HVn, h = bh % HVn;
  int kh = h >> 1;                     // rep = HV/HK = 2 (repeat_interleave)
  int t = threadIdx.x;
  const float* qp = qn + ((size_t)(b * HKn + kh)) * SS * DKc;
  const float* kp = kn + ((size_t)(b * HKn + kh)) * SS * DKc;
  const float* vp = vt + ((size_t)(b * HVn + h))  * SS * DVc;
  float St[64];
  #pragma unroll
  for (int i = 0; i < 64; ++i) St[i] = 0.f;
  __shared__ float lk[64], lq[64];
  for (int s = 0; s < SS; ++s) {
    float g  = gbuf[((size_t)(b * SS + s)) * HVn + h];
    float be = bbuf[((size_t)(b * SS + s)) * HVn + h];
    lk[t] = kp[(size_t)s * DKc + t];
    lq[t] = qp[(size_t)s * DKc + t];
    float v = vp[(size_t)s * DVc + t];
    __syncthreads();
    float dec = __expf(g);
    float kv = 0.f;
    #pragma unroll
    for (int i = 0; i < 64; ++i) { St[i] *= dec; kv += lk[i] * St[i]; }
    float delta = (v - kv) * be;
    float o = 0.f;
    #pragma unroll
    for (int i = 0; i < 64; ++i) { St[i] += lk[i] * delta; o += lq[i] * St[i]; }
    obuf[((size_t)(b * SS + s)) * VDIMc + h * DVc + t] = o;
    __syncthreads();
  }
}

// ---------------- RMSNorm + (1+w) + SiLU(z) gate -> bf16 ----------------
__global__ __launch_bounds__(256) void gate_norm_kernel(const float* __restrict__ obuf,
    const float* __restrict__ zbuf, const float* __restrict__ normw,
    __bf16* __restrict__ og) {
  int row = blockIdx.x;
  int t = threadIdx.x;
  int d = t & 63;
  __shared__ float ss[HVn];
  if (t < HVn) ss[t] = 0.f;
  __syncthreads();
  float vals[4];
  #pragma unroll
  for (int p = 0; p < 4; ++p) {
    int h = p * 4 + (t >> 6);
    float o = obuf[(size_t)row * VDIMc + h * DVc + d];
    vals[p] = o;
    atomicAdd(&ss[h], o * o);
  }
  __syncthreads();
  #pragma unroll
  for (int p = 0; p < 4; ++p) {
    int h = p * 4 + (t >> 6);
    float ms = ss[h] * (1.f / DVc);
    float o  = vals[p] * rsqrtf(ms + EPSc) * (1.f + normw[d]);
    float z  = zbuf[(size_t)row * VDIMc + h * DVc + d];
    o *= z / (1.f + __expf(-z));       // silu gate
    og[(size_t)row * VDIMc + h * DVc + d] = f2bf(o);
  }
}

// ---------------- host orchestration ----------------
extern "C" void kernel_launch(void* const* d_in, const int* in_sizes, int n_in,
                              void* d_out, int out_size, void* d_ws, size_t ws_size,
                              hipStream_t stream) {
  (void)in_sizes; (void)n_in; (void)out_size; (void)ws_size;
  const float* x     = (const float*)d_in[0];
  const float* Wqkv  = (const float*)d_in[1];
  const float* convw = (const float*)d_in[2];
  const float* Wz    = (const float*)d_in[3];
  const float* Wb    = (const float*)d_in[4];
  const float* Wa    = (const float*)d_in[5];
  const float* dtb   = (const float*)d_in[6];
  const float* Alog  = (const float*)d_in[7];
  const float* normw = (const float*)d_in[8];
  const float* Wout  = (const float*)d_in[9];
  float* out = (float*)d_out;

  char* ws = (char*)d_ws;
  size_t off = 0;
  auto alloc = [&](size_t bytes) -> void* {
    void* p = ws + off;
    off += (bytes + 255) & ~(size_t)255;
    return p;
  };
  __bf16* x_bf    = (__bf16*)alloc((size_t)MMr * DD * 2);
  __bf16* wqkv_bf = (__bf16*)alloc((size_t)CDIMc * DD * 2);
  __bf16* wz_bf   = (__bf16*)alloc((size_t)VDIMc * DD * 2);
  __bf16* wout_bf = (__bf16*)alloc((size_t)DD * VDIMc * 2);
  float*  qkv_pre = (float*)alloc((size_t)MMr * CDIMc * 4);
  float*  mixed   = (float*)alloc((size_t)MMr * CDIMc * 4);
  float*  zbuf    = (float*)alloc((size_t)MMr * VDIMc * 4);
  float*  qn      = (float*)alloc((size_t)BB * HKn * SS * DKc * 4);
  float*  kn      = (float*)alloc((size_t)BB * HKn * SS * DKc * 4);
  float*  vt      = (float*)alloc((size_t)BB * HVn * SS * DVc * 4);
  float*  beta    = (float*)alloc((size_t)MMr * HVn * 4);
  float*  gbuf    = (float*)alloc((size_t)MMr * HVn * 4);
  float*  obuf    = (float*)alloc((size_t)MMr * VDIMc * 4);
  __bf16* og_bf   = (__bf16*)alloc((size_t)MMr * VDIMc * 2);

  // casts to bf16
  cast_f32_bf16<<<(MMr*DD/4 + 255)/256, 256, 0, stream>>>(x, x_bf, MMr*DD/4);
  cast_f32_bf16<<<(CDIMc*DD/4 + 255)/256, 256, 0, stream>>>(Wqkv, wqkv_bf, CDIMc*DD/4);
  cast_f32_bf16<<<(VDIMc*DD/4 + 255)/256, 256, 0, stream>>>(Wz, wz_bf, VDIMc*DD/4);
  cast_f32_bf16<<<(DD*VDIMc/4 + 255)/256, 256, 0, stream>>>(Wout, wout_bf, DD*VDIMc/4);

  // small projections (beta, g)
  proj_bg_kernel<<<MMr, 32, 0, stream>>>(x, Wb, Wa, dtb, Alog, beta, gbuf);

  // qkv = x @ Wqkv^T  [4096 x 2048], z = x @ Wz^T [4096 x 1024]  (WMMA bf16)
  wmma_gemm_bf16<<<dim3(MMr/64, CDIMc/128), 256, 0, stream>>>(x_bf, wqkv_bf, qkv_pre, MMr, CDIMc, DD);
  wmma_gemm_bf16<<<dim3(MMr/64, VDIMc/128), 256, 0, stream>>>(x_bf, wz_bf, zbuf, MMr, VDIMc, DD);

  // depthwise causal conv + silu
  conv_silu_kernel<<<(size_t)MMr*CDIMc/256, 256, 0, stream>>>(qkv_pre, convw, mixed);

  // l2norm q/k + transpose into scan layouts
  scanprep_kernel<<<MMr, 256, 0, stream>>>(mixed, qn, kn, vt);

  // sequential delta-rule scan: one block per (b,h)
  scan_kernel<<<BB*HVn, 64, 0, stream>>>(qn, kn, vt, gbuf, beta, obuf);

  // RMSNorm + gate -> bf16
  gate_norm_kernel<<<MMr, 256, 0, stream>>>(obuf, zbuf, normw, og_bf);

  // out = og @ Wout^T  [4096 x 1024]  (WMMA bf16)
  wmma_gemm_bf16<<<dim3(MMr/64, DD/128), 256, 0, stream>>>(og_bf, wout_bf, out, MMr, DD, VDIMc);
}
